// NetGINE_28432683499894
// MI455X (gfx1250) — matrix operand order repaired and checked by reading, not compiled, not run on hardware
//
#include <hip/hip_runtime.h>
#include <hip/hip_bf16.h>

#define N_NODES   80000
#define N_EDGES   1280000
#define N_GRAPHS  256
#define EDGE_FEAT 16
#define DIM       64
#define N_LAYERS  4

typedef __bf16 bf16;
typedef __attribute__((ext_vector_type(16))) __bf16 v16bf;
typedef __attribute__((ext_vector_type(8)))  __bf16 v8bf;
typedef __attribute__((ext_vector_type(4)))  __bf16 v4bf;
typedef __attribute__((ext_vector_type(8)))  float  v8f;
typedef __attribute__((ext_vector_type(4)))  float  v4f;

__device__ __forceinline__ v8f zero_v8f() {
  v8f z = {0.f, 0.f, 0.f, 0.f, 0.f, 0.f, 0.f, 0.f};
  return z;
}

// D = A(16x32 bf16) x B(32x16 bf16) + C (f32)
__device__ __forceinline__ v8f wmma_bf16(v16bf a, v16bf b, v8f c) {
  return __builtin_amdgcn_wmma_f32_16x16x32_bf16(
      /*neg_a=*/false, a, /*neg_b=*/false, b,
      /*c_mod=*/(short)0, c, /*reuse_a=*/false, /*reuse_b=*/false);
}

// wave-local ordering of LDS stage-write -> cross-lane stage-read
__device__ __forceinline__ void lds_fence() {
  asm volatile("s_wait_dscnt 0" ::: "memory");
}

// B fragment (32x16) from LDS weight matrix W (rows = K, 64 cols, row-major).
// Lanes 0-15: col = lane, K = kbase+0..7,16..23 ; lanes 16-31: col = lane-16, K = kbase+8..15,24..31.
__device__ __forceinline__ v16bf make_b_frag(const bf16* W, int kbase, int ntile, int lane) {
  int col = ntile * 16 + (lane & 15);
  int k0  = kbase + (lane >> 4) * 8;
  v16bf b;
#pragma unroll
  for (int j = 0; j < 16; ++j) b[j] = W[(k0 + j + (j >= 8 ? 8 : 0)) * DIM + col];
  return b;
}

// A fragment (16x32) from a 16x64 bf16 LDS tile (two contiguous 16B runs per lane).
// Lanes 0-15: row = lane, elems 0..7 = K kbase+0..7, elems 8..15 = K kbase+16..23.
// Lanes 16-31: row = lane-16, elems 0..7 = K kbase+8..15, elems 8..15 = K kbase+24..31.
__device__ __forceinline__ v16bf make_a_frag(const bf16* T, int kbase, int lane) {
  int r  = lane & 15;
  int kh = (lane >> 4) * 8;
  const v8bf* p = (const v8bf*)(T + r * DIM + kbase + kh);  // 16B aligned
  v8bf lo = p[0];
  v8bf hi = p[2];   // +16 bf16 elements
  return __builtin_shufflevector(lo, hi, 0, 1, 2, 3, 4, 5, 6, 7,
                                 8, 9, 10, 11, 12, 13, 14, 15);
}

// ---------------- fused edge pipeline ----------------
// e = relu(EA@W1+b1)@W2+b2 ; m = relu(x[src]+e)*ew ; atomic scatter-add into agg[dst]
__global__ __launch_bounds__(128) void gine_edge_kernel(
    const float* __restrict__ x, const float* __restrict__ ea,
    const float* __restrict__ ew, const int* __restrict__ ei,
    const float* __restrict__ w1, const float* __restrict__ b1,
    const float* __restrict__ w2, const float* __restrict__ b2,
    float* __restrict__ agg)
{
  __shared__ __align__(16) bf16  sW1[32 * DIM];      // K padded 16 -> 32 with zeros
  __shared__ __align__(16) bf16  sW2[DIM * DIM];
  __shared__ __align__(16) bf16  sT[4][16 * DIM];    // per-wave staging tile
  __shared__ float sB1[DIM], sB2[DIM];

  int tid = threadIdx.x;
  for (int i = tid; i < 32 * DIM; i += 128)
    sW1[i] = (i < EDGE_FEAT * DIM) ? (bf16)w1[i] : (bf16)0.0f;
  for (int i = tid; i < DIM * DIM; i += 128)
    sW2[i] = (bf16)w2[i];
  if (tid < DIM) { sB1[tid] = b1[tid]; sB2[tid] = b2[tid]; }
  __syncthreads();

  int wave = tid >> 5, lane = tid & 31;
  bf16* T = sT[wave];
  const int* srcI = ei;
  const int* dstI = ei + N_EDGES;

  v16bf bw1[4], bw2[2][4];
#pragma unroll
  for (int n = 0; n < 4; ++n) bw1[n] = make_b_frag(sW1, 0, n, lane);
#pragma unroll
  for (int c = 0; c < 2; ++c)
#pragma unroll
    for (int n = 0; n < 4; ++n) bw2[c][n] = make_b_frag(sW2, c * 32, n, lane);

  int r  = lane & 15;
  int kh = (lane >> 4) * 8;
  int mb = (lane >> 4) * 8;
  const int nTiles = N_EDGES / 16;   // exact

  for (int t = blockIdx.x * 4 + wave; t < nTiles; t += gridDim.x * 4) {
    int e0 = t * 16;

    // A fragment directly from edge_attr (K=16 padded to 32), two 16B loads
    const v4f* ep4 = (const v4f*)(ea + (size_t)(e0 + r) * EDGE_FEAT + kh);
    v4f ev0 = ep4[0], ev1 = ep4[1];
    v16bf aE;
#pragma unroll
    for (int j = 0; j < 4; ++j) {
      aE[j]      = (bf16)ev0[j];
      aE[j + 4]  = (bf16)ev1[j];
      aE[j + 8]  = (bf16)0.0f;
      aE[j + 12] = (bf16)0.0f;
    }

    v8f acc[4];
#pragma unroll
    for (int n = 0; n < 4; ++n) acc[n] = wmma_bf16(aE, bw1[n], zero_v8f());

    // bias + relu -> bf16 staging tile
#pragma unroll
    for (int n = 0; n < 4; ++n) {
      int col = n * 16 + (lane & 15);
      float bias = sB1[col];
#pragma unroll
      for (int q = 0; q < 8; ++q)
        T[(mb + q) * DIM + col] = (bf16)fmaxf(acc[n][q] + bias, 0.f);
    }
    lds_fence();

    v8f acc2[4];
#pragma unroll
    for (int n = 0; n < 4; ++n) acc2[n] = zero_v8f();
#pragma unroll
    for (int c = 0; c < 2; ++c) {
      v16bf aT = make_a_frag(T, c * 32, lane);
#pragma unroll
      for (int n = 0; n < 4; ++n) acc2[n] = wmma_bf16(aT, bw2[c][n], acc2[n]);
    }

    // per-lane edge metadata for its 8 rows
    int si[8], di[8]; float wgt[8];
#pragma unroll
    for (int q = 0; q < 8; ++q) {
      int e = e0 + mb + q;
      si[q] = srcI[e]; di[q] = dstI[e]; wgt[q] = ew[e];
    }

    // message + scatter: coalesced 64B per 16-lane group
#pragma unroll
    for (int n = 0; n < 4; ++n) {
      int col = n * 16 + (lane & 15);
      float bias = sB2[col];
#pragma unroll
      for (int q = 0; q < 8; ++q) {
        float evv = acc2[n][q] + bias;
        float xv  = x[(size_t)si[q] * DIM + col];
        float mv  = fmaxf(xv + evv, 0.f) * wgt[q];
        atomicAdd(&agg[(size_t)di[q] * DIM + col], mv);
      }
    }
  }
}

// ---------------- node MLP + BN partial sums ----------------
__global__ __launch_bounds__(128) void gine_node_kernel(
    const float* __restrict__ x, const float* __restrict__ agg,
    const float* __restrict__ epsArr, int layer,
    const float* __restrict__ w1, const float* __restrict__ b1,
    const float* __restrict__ w2, const float* __restrict__ b2,
    float* __restrict__ hout, float* __restrict__ gSum, float* __restrict__ gSq)
{
  __shared__ __align__(16) bf16  sW1[DIM * DIM];
  __shared__ __align__(16) bf16  sW2[DIM * DIM];
  __shared__ __align__(16) bf16  sT[4][16 * DIM];
  __shared__ float sB1[DIM], sB2[DIM];
  __shared__ float sSum[DIM], sSq[DIM];

  int tid = threadIdx.x;
  for (int i = tid; i < DIM * DIM; i += 128) {
    sW1[i] = (bf16)w1[i];
    sW2[i] = (bf16)w2[i];
  }
  if (tid < DIM) { sB1[tid] = b1[tid]; sB2[tid] = b2[tid]; sSum[tid] = 0.f; sSq[tid] = 0.f; }
  __syncthreads();

  float epv = 1.0f + epsArr[layer];
  int wave = tid >> 5, lane = tid & 31;
  bf16* T = sT[wave];

  v16bf bw1[2][4], bw2[2][4];
#pragma unroll
  for (int c = 0; c < 2; ++c)
#pragma unroll
    for (int n = 0; n < 4; ++n) {
      bw1[c][n] = make_b_frag(sW1, c * 32, n, lane);
      bw2[c][n] = make_b_frag(sW2, c * 32, n, lane);
    }

  int mb = (lane >> 4) * 8;
  const int nTiles = N_NODES / 16;   // exact

  for (int t = blockIdx.x * 4 + wave; t < nTiles; t += gridDim.x * 4) {
    size_t base = (size_t)t * 16 * DIM;

    // stage h0 = (1+eps)*x + agg as bf16 (vectorized: 2x b128 load + b64 LDS store)
#pragma unroll
    for (int j = 0; j < 8; ++j) {
      int i = j * 128 + lane * 4;
      v4f xv = *(const v4f*)(x + base + i);
      v4f av = *(const v4f*)(agg + base + i);
      v4bf o;
#pragma unroll
      for (int k = 0; k < 4; ++k) o[k] = (bf16)fmaf(epv, xv[k], av[k]);
      *(v4bf*)(T + i) = o;
    }
    lds_fence();

    v8f acc[4];
#pragma unroll
    for (int n = 0; n < 4; ++n) acc[n] = zero_v8f();
#pragma unroll
    for (int c = 0; c < 2; ++c) {
      v16bf aT = make_a_frag(T, c * 32, lane);
#pragma unroll
      for (int n = 0; n < 4; ++n) acc[n] = wmma_bf16(aT, bw1[c][n], acc[n]);
    }
    lds_fence();

#pragma unroll
    for (int n = 0; n < 4; ++n) {
      int col = n * 16 + (lane & 15);
      float bias = sB1[col];
#pragma unroll
      for (int q = 0; q < 8; ++q)
        T[(mb + q) * DIM + col] = (bf16)fmaxf(acc[n][q] + bias, 0.f);
    }
    lds_fence();

    v8f acc2[4];
#pragma unroll
    for (int n = 0; n < 4; ++n) acc2[n] = zero_v8f();
#pragma unroll
    for (int c = 0; c < 2; ++c) {
      v16bf aT = make_a_frag(T, c * 32, lane);
#pragma unroll
      for (int n = 0; n < 4; ++n) acc2[n] = wmma_bf16(aT, bw2[c][n], acc2[n]);
    }

#pragma unroll
    for (int n = 0; n < 4; ++n) {
      int col = n * 16 + (lane & 15);
      float bias = sB2[col];
      float ps = 0.f, pq = 0.f;
#pragma unroll
      for (int q = 0; q < 8; ++q) {
        float v = acc2[n][q] + bias;
        hout[base + (size_t)(mb + q) * DIM + col] = v;
        ps += v; pq += v * v;
      }
      atomicAdd(&sSum[col], ps);
      atomicAdd(&sSq[col], pq);
    }
  }
  __syncthreads();
  if (tid < DIM) { atomicAdd(&gSum[tid], sSum[tid]); atomicAdd(&gSq[tid], sSq[tid]); }
}

// ---------------- BN finalize ----------------
__global__ void gine_bnstat_kernel(const float* __restrict__ gSum, const float* __restrict__ gSq,
                                   const float* __restrict__ gamma, const float* __restrict__ beta,
                                   int layer, float* __restrict__ scale, float* __restrict__ shift)
{
  int c = threadIdx.x;
  if (c < DIM) {
    const float invN = 1.0f / (float)N_NODES;
    float mu  = gSum[c] * invN;
    float var = gSq[c] * invN - mu * mu;
    float sc  = gamma[layer * DIM + c] * rsqrtf(var + 1e-5f);
    scale[c] = sc;
    shift[c] = beta[layer * DIM + c] - mu * sc;
  }
}

// BN apply + ReLU -> x_out ; re-zero agg & stats for next layer (float4 path)
__global__ void gine_bnrelu_kernel(const float* __restrict__ h,
                                   const float* __restrict__ scale,
                                   const float* __restrict__ shift,
                                   float* __restrict__ xout,
                                   float* __restrict__ agg,
                                   float* __restrict__ stats /* 128 floats */)
{
  if (blockIdx.x == 0 && threadIdx.x < 128) stats[threadIdx.x] = 0.f;
  const v4f* h4 = (const v4f*)h;
  v4f* x4 = (v4f*)xout;
  v4f* a4 = (v4f*)agg;
  const size_t n4 = (size_t)N_NODES * DIM / 4;
  const v4f z = {0.f, 0.f, 0.f, 0.f};
  for (size_t i = (size_t)blockIdx.x * blockDim.x + threadIdx.x; i < n4;
       i += (size_t)gridDim.x * blockDim.x) {
    int cb = (int)(i & 15) * 4;
    v4f hv = h4[i], o;
#pragma unroll
    for (int k = 0; k < 4; ++k)
      o[k] = fmaxf(fmaf(hv[k], scale[cb + k], shift[cb + k]), 0.f);
    x4[i] = o;
    a4[i] = z;
  }
}

__global__ void gine_zero_kernel(float* __restrict__ p, int n /* multiple of 4 */) {
  v4f* p4 = (v4f*)p;
  int n4 = n >> 2;
  const v4f z = {0.f, 0.f, 0.f, 0.f};
  for (int i = blockIdx.x * blockDim.x + threadIdx.x; i < n4; i += gridDim.x * blockDim.x)
    p4[i] = z;
}

// ---------------- pooling ----------------
__global__ void gine_pool_kernel(const float* __restrict__ x, const int* __restrict__ batch,
                                 float* __restrict__ gsum, float* __restrict__ counts)
{
  const v4f* x4 = (const v4f*)x;
  const size_t n4 = (size_t)N_NODES * DIM / 4;
  for (size_t i = (size_t)blockIdx.x * blockDim.x + threadIdx.x; i < n4;
       i += (size_t)gridDim.x * blockDim.x) {
    int node = (int)(i >> 4), cb = (int)(i & 15) * 4;
    int g = batch[node];
    v4f v = x4[i];
#pragma unroll
    for (int k = 0; k < 4; ++k)
      atomicAdd(&gsum[(size_t)g * DIM + cb + k], v[k]);
    if (cb == 0) atomicAdd(&counts[g], 1.0f);
  }
}

// ---------------- readout head: 3x(64x64)+relu then 64->1 ----------------
__global__ __launch_bounds__(64) void gine_head_kernel(
    const float* __restrict__ gsum, const float* __restrict__ counts,
    const float* __restrict__ w1, const float* __restrict__ b1,
    const float* __restrict__ w2, const float* __restrict__ b2,
    const float* __restrict__ w3, const float* __restrict__ b3,
    const float* __restrict__ w4, const float* __restrict__ b4,
    float* __restrict__ out)
{
  __shared__ float A[DIM], B[DIM];
  int g = blockIdx.x, c = threadIdx.x;
  float cnt = fmaxf(counts[g], 1.0f);
  A[c] = gsum[(size_t)g * DIM + c] / cnt;
  __syncthreads();

  float acc = b1[c];
  for (int k = 0; k < DIM; ++k) acc = fmaf(A[k], w1[k * DIM + c], acc);
  B[c] = fmaxf(acc, 0.f);
  __syncthreads();

  acc = b2[c];
  for (int k = 0; k < DIM; ++k) acc = fmaf(B[k], w2[k * DIM + c], acc);
  A[c] = fmaxf(acc, 0.f);
  __syncthreads();

  acc = b3[c];
  for (int k = 0; k < DIM; ++k) acc = fmaf(A[k], w3[k * DIM + c], acc);
  B[c] = fmaxf(acc, 0.f);
  __syncthreads();

  A[c] = B[c] * w4[c];
  __syncthreads();
  if (c == 0) {
    float s = 0.f;
    for (int k = 0; k < DIM; ++k) s += A[k];
    out[g] = s + b4[0];
  }
}

extern "C" void kernel_launch(void* const* d_in, const int* in_sizes, int n_in,
                              void* d_out, int out_size, void* d_ws, size_t ws_size,
                              hipStream_t stream)
{
  const float* x_in  = (const float*)d_in[0];
  const float* ea    = (const float*)d_in[1];
  const float* ew    = (const float*)d_in[2];
  const int*   ei    = (const int*)d_in[3];
  const int*   batch = (const int*)d_in[4];
  const float* bw1   = (const float*)d_in[5];
  const float* bb1   = (const float*)d_in[6];
  const float* bw2   = (const float*)d_in[7];
  const float* bb2   = (const float*)d_in[8];
  const float* mw1   = (const float*)d_in[9];
  const float* mb1   = (const float*)d_in[10];
  const float* mw2   = (const float*)d_in[11];
  const float* mb2   = (const float*)d_in[12];
  const float* eps   = (const float*)d_in[13];
  const float* gam   = (const float*)d_in[14];
  const float* bet   = (const float*)d_in[15];
  const float* f1w   = (const float*)d_in[16];
  const float* f1b   = (const float*)d_in[17];
  const float* f2w   = (const float*)d_in[18];
  const float* f2b   = (const float*)d_in[19];
  const float* f3w   = (const float*)d_in[20];
  const float* f3b   = (const float*)d_in[21];
  const float* f4w   = (const float*)d_in[22];
  const float* f4b   = (const float*)d_in[23];
  float* out = (float*)d_out;

  // workspace layout
  const size_t NF = (size_t)N_NODES * DIM;
  float* W    = (float*)d_ws;
  float* agg  = W;                    // NF
  float* hbuf = agg + NF;             // NF
  float* xbuf = hbuf + NF;            // NF
  float* gS   = xbuf + NF;            // 64
  float* gQ   = gS + 64;              // 64 (contiguous with gS -> 128 stats)
  float* scl  = gQ + 64;              // 64
  float* shf  = scl + 64;             // 64
  float* pg   = shf + 64;             // 256*64
  float* pc   = pg + (size_t)N_GRAPHS * DIM; // 256 (contiguous with pg)

  gine_zero_kernel<<<2048, 256, 0, stream>>>(agg, (int)NF);
  gine_zero_kernel<<<1, 128, 0, stream>>>(gS, 128);

  for (int l = 0; l < N_LAYERS; ++l) {
    const float* xcur = (l == 0) ? x_in : xbuf;
    gine_edge_kernel<<<640, 128, 0, stream>>>(
        xcur, ea, ew, ei,
        bw1 + (size_t)l * EDGE_FEAT * DIM, bb1 + l * DIM,
        bw2 + (size_t)l * DIM * DIM,       bb2 + l * DIM, agg);
    gine_node_kernel<<<256, 128, 0, stream>>>(
        xcur, agg, eps, l,
        mw1 + (size_t)l * DIM * DIM, mb1 + l * DIM,
        mw2 + (size_t)l * DIM * DIM, mb2 + l * DIM,
        hbuf, gS, gQ);
    gine_bnstat_kernel<<<1, 64, 0, stream>>>(gS, gQ, gam, bet, l, scl, shf);
    gine_bnrelu_kernel<<<2048, 256, 0, stream>>>(hbuf, scl, shf, xbuf, agg, gS);
  }

  gine_zero_kernel<<<64, 256, 0, stream>>>(pg, N_GRAPHS * DIM + N_GRAPHS);
  gine_pool_kernel<<<2048, 256, 0, stream>>>(xbuf, batch, pg, pc);
  gine_head_kernel<<<N_GRAPHS, 64, 0, stream>>>(pg, pc, f1w, f1b, f2w, f2b,
                                                f3w, f3b, f4w, f4b, out);

  (void)in_sizes; (void)n_in; (void)out_size; (void)ws_size;
}
